// DiffHead_1348619731551
// MI455X (gfx1250) — matrix operand order repaired and checked by reading, not compiled, not run on hardware
//
#include <hip/hip_runtime.h>

// ---------- types ----------
typedef __attribute__((ext_vector_type(16))) __bf16 v16bf;
typedef __attribute__((ext_vector_type(8)))  float  v8f;

union V16 { v16bf v; uint4 q[2]; };   // 32 bytes: 16 bf16 halves
static_assert(sizeof(V16) == 32, "V16 size");

#define BATCH 4
#define SEQ   2048
#define EMBD  1024
#define HEADD 64
#define VOUT  128
#define NTOK  (BATCH * SEQ)
#define NCOL  384            // 4*64 + 128 projected columns

__device__ __forceinline__ unsigned short f2bf_bits(float f) {
    union { float f; unsigned u; } u; u.f = f;
    unsigned r = u.u + 0x7FFFu + ((u.u >> 16) & 1u);   // round-to-nearest-even
    return (unsigned short)(r >> 16);
}

__device__ __forceinline__ unsigned pk2(unsigned short a, unsigned short b) {
    return (unsigned)a | ((unsigned)b << 16);
}

// exchange with lane^16 via ds_swizzle SWAPX16 (xor=0x10, or=0, and=0x1f)
__device__ __forceinline__ float xor16(float v) {
    return __int_as_float(__builtin_amdgcn_ds_swizzle(__float_as_int(v), 0x401F));
}

__device__ __forceinline__ v8f wmma_bf16(const V16& a, const V16& b, v8f c) {
    return __builtin_amdgcn_wmma_f32_16x16x32_bf16(false, a.v, false, b.v,
                                                   (short)0, c, false, false);
}

// ---------------------------------------------------------------------------
// x (f32) -> xb (bf16), coalesced
// ---------------------------------------------------------------------------
__global__ __launch_bounds__(256) void xcvt_kernel(const float* __restrict__ x,
                                                   unsigned short* __restrict__ xb)
{
    const size_t i = ((size_t)blockIdx.x * 256 + threadIdx.x) * 4;
    const float4 v = *(const float4*)(x + i);
    ushort4 o;
    o.x = f2bf_bits(v.x); o.y = f2bf_bits(v.y);
    o.z = f2bf_bits(v.z); o.w = f2bf_bits(v.w);
    *(ushort4*)(xb + i) = o;
}

// ---------------------------------------------------------------------------
// Pack + transpose the 5 weight matrices into Wt[384][1024] bf16
// cols: 0-63 Wq1 | 64-127 Wk1 | 128-191 Wq2 | 192-255 Wk2 | 256-383 Wv
// ---------------------------------------------------------------------------
__global__ __launch_bounds__(256) void wcvt_kernel(
    const float* __restrict__ Wq1, const float* __restrict__ Wk1,
    const float* __restrict__ Wq2, const float* __restrict__ Wk2,
    const float* __restrict__ Wv,  unsigned short* __restrict__ Wt)
{
    const int col = blockIdx.x;                       // 0..383
    const int k   = blockIdx.y * 256 + threadIdx.x;   // 0..1023
    const float* W; int wc, stride;
    if (col < 64)       { W = Wq1; wc = col;       stride = 64;  }
    else if (col < 128) { W = Wk1; wc = col - 64;  stride = 64;  }
    else if (col < 192) { W = Wq2; wc = col - 128; stride = 64;  }
    else if (col < 256) { W = Wk2; wc = col - 192; stride = 64;  }
    else                { W = Wv;  wc = col - 256; stride = 128; }
    Wt[(size_t)col * EMBD + k] = f2bf_bits(W[(size_t)k * stride + wc]);
}

// ---------------------------------------------------------------------------
// Projection GEMM via WMMA: [8192 x 1024] @ [1024 x 384]
// wave owns 32 tokens x 64 cols (2x4 tiles); k-loop is double-buffered so the
// next A/B tiles are in flight while the current 8 WMMAs execute.
// outputs: q1,k1,q2,k2 row-major [t][64]; v -> Vt[b][col][T] (transposed)
// ---------------------------------------------------------------------------
__global__ __launch_bounds__(128, 1) void proj_wmma_kernel(
    const unsigned short* __restrict__ xb,   // [NTOK][1024]
    const unsigned short* __restrict__ Wt,   // [384][1024]
    unsigned short* __restrict__ q1, unsigned short* __restrict__ k1,
    unsigned short* __restrict__ q2, unsigned short* __restrict__ k2,
    unsigned short* __restrict__ Vt)
{
    const int lane = threadIdx.x & 31;
    const int ln   = lane & 15;
    const int kh   = lane >> 4;
    const int wid  = blockIdx.x * 4 + (threadIdx.x >> 5);   // 0..1535
    const int mg   = wid / 6;                               // 32-token group
    const int ng   = wid - mg * 6;                          // 64-col group
    const int t0   = mg * 32;
    const int c0   = ng * 64;

    const unsigned short* xr0 = xb + (size_t)(t0 + ln) * EMBD;
    const unsigned short* xr1 = xr0 + 16 * EMBD;
    const unsigned short* wbase[4];
#pragma unroll
    for (int nt = 0; nt < 4; ++nt)
        wbase[nt] = Wt + (size_t)(c0 + nt * 16 + ln) * EMBD;

    const v8f vz = {0.f,0.f,0.f,0.f,0.f,0.f,0.f,0.f};
    v8f acc[2][4];
#pragma unroll
    for (int m = 0; m < 2; ++m)
#pragma unroll
        for (int nt = 0; nt < 4; ++nt) acc[m][nt] = vz;

    auto loadA = [&](int k, V16* a2) {
        a2[0].q[0] = *(const uint4*)(xr0 + k + 8 * kh);
        a2[0].q[1] = *(const uint4*)(xr0 + k + 16 + 8 * kh);
        a2[1].q[0] = *(const uint4*)(xr1 + k + 8 * kh);
        a2[1].q[1] = *(const uint4*)(xr1 + k + 16 + 8 * kh);
    };
    auto loadB = [&](int k, V16* b4) {
#pragma unroll
        for (int nt = 0; nt < 4; ++nt) {
            const unsigned short* w = wbase[nt] + k + 8 * kh;
            b4[nt].q[0] = *(const uint4*)(w);
            b4[nt].q[1] = *(const uint4*)(w + 16);
        }
    };
    auto mma8 = [&](V16* a2, V16* b4) {
#pragma unroll
        for (int nt = 0; nt < 4; ++nt) {
            acc[0][nt] = wmma_bf16(a2[0], b4[nt], acc[0][nt]);
            acc[1][nt] = wmma_bf16(a2[1], b4[nt], acc[1][nt]);
        }
    };

    V16 aA[2], bA[4], aB[2], bB[4];
    loadA(0, aA); loadB(0, bA);
    for (int k0 = 0; k0 < EMBD; k0 += 64) {
        loadA(k0 + 32, aB); loadB(k0 + 32, bB);      // next tile in flight
        mma8(aA, bA);
        if (k0 + 64 < EMBD) { loadA(k0 + 64, aA); loadB(k0 + 64, bA); }
        mma8(aB, bB);
    }

    unsigned short* dst = q1;
    if (ng == 1) dst = k1; else if (ng == 2) dst = q2; else if (ng == 3) dst = k2;
#pragma unroll
    for (int m = 0; m < 2; ++m)
#pragma unroll
    for (int nt = 0; nt < 4; ++nt)
#pragma unroll
    for (int r = 0; r < 8; ++r) {
        const int t = t0 + m * 16 + r + 8 * kh;
        const unsigned short bits = f2bf_bits(acc[m][nt][r]);
        if (ng < 4) {
            dst[(size_t)t * HEADD + nt * 16 + ln] = bits;
        } else {
            const int vc = c0 - 256 + nt * 16 + ln;
            const int bb = t >> 11, tl = t & (SEQ - 1);
            Vt[((size_t)bb * VOUT + vc) * SEQ + tl] = bits;
        }
    }
}

// ---------------------------------------------------------------------------
// lambda scalar: lam = mean(exp(lq1*lk1) - exp(lq2*lk2)) + dyn_init
// ---------------------------------------------------------------------------
__global__ void lam_kernel(const float* __restrict__ lq1, const float* __restrict__ lk1,
                           const float* __restrict__ lq2, const float* __restrict__ lk2,
                           const int* __restrict__ lidx, float* __restrict__ lamOut)
{
    const int j = threadIdx.x;            // 32 threads
    float d = 0.f;
    for (int jj = j; jj < HEADD; jj += 32)
        d += __expf(lq1[jj] * lk1[jj]) - __expf(lq2[jj] * lk2[jj]);
#pragma unroll
    for (int off = 16; off > 0; off >>= 1) d += __shfl_xor(d, off, 32);
    if (j == 0) {
        float li  = (float)lidx[0];
        float dyn = 0.8f - 0.6f * __expf(-0.3f * (li - 1.0f));
        lamOut[0] = d / 64.0f + dyn;
    }
}

// ---------------------------------------------------------------------------
// Flash attention, transposed dataflow (no LDS), both heads concurrently:
//   S^T = K_tile x Q^T ; O^T = V^T x P^T (P^T built with one SWAPX16/reg)
// Pipeline per key tile: S-WMMAs -> issue all V loads + next-K loads ->
// softmax VALU (hides load latency) -> PV WMMAs.
// blockIdx.y selects head; O1/O2 written to workspace, combined later.
// ---------------------------------------------------------------------------
__global__ __launch_bounds__(128, 1) void attn_kernel(
    const unsigned short* __restrict__ Q1, const unsigned short* __restrict__ K1,
    const unsigned short* __restrict__ Q2, const unsigned short* __restrict__ K2,
    const unsigned short* __restrict__ Vt,
    float* __restrict__ Obuf)                 // [2][B*T][128]
{
    const int head = blockIdx.y;
    const unsigned short* Q  = head ? Q2 : Q1;
    const unsigned short* Km = head ? K2 : K1;
    float* out = Obuf + (size_t)head * NTOK * VOUT;

    const int lane = threadIdx.x & 31;
    const int wv   = threadIdx.x >> 5;
    const int ln   = lane & 15;
    const int kh   = lane >> 4;

    const int tile   = blockIdx.x * 4 + wv;   // 0..511
    const int b      = tile >> 7;
    const int qt     = tile & 127;
    const int t_base = qt * 16;

    const float SCALE = 0.125f;               // 1/sqrt(64)

    // Q^T as B operand (lane = column t, halves = table-order d)
    V16 qb[2];
    {
        const unsigned short* qrow = Q + (size_t)(b * SEQ + t_base + ln) * HEADD;
#pragma unroll
        for (int c2 = 0; c2 < 2; ++c2) {
            qb[c2].q[0] = *(const uint4*)(qrow + 32 * c2 + 8 * kh);
            qb[c2].q[1] = *(const uint4*)(qrow + 32 * c2 + 16 + 8 * kh);
        }
    }

    const v8f vz = {0.f,0.f,0.f,0.f,0.f,0.f,0.f,0.f};
    v8f accT[8];
#pragma unroll
    for (int cb = 0; cb < 8; ++cb) accT[cb] = vz;

    float mrun = -1e30f, lrun = 0.f;          // per-lane (t = t_base + ln)
    const int tg  = t_base + ln;
    const int nkt = (t_base + 15) / 32 + 1;

    auto loadK = [&](int sb, V16* k4) {
        const unsigned short* kr0 = Km + (size_t)(b * SEQ + sb + ln) * HEADD;
        const unsigned short* kr1 = kr0 + 16 * HEADD;
#pragma unroll
        for (int c2 = 0; c2 < 2; ++c2) {
            const unsigned short* p0 = kr0 + 32 * c2 + 8 * kh;
            const unsigned short* p1 = kr1 + 32 * c2 + 8 * kh;
            k4[c2].q[0]     = *(const uint4*)(p0);
            k4[c2].q[1]     = *(const uint4*)(p0 + 16);
            k4[2 + c2].q[0] = *(const uint4*)(p1);
            k4[2 + c2].q[1] = *(const uint4*)(p1 + 16);
        }
    };

    auto step = [&](int kt, V16* kc, V16* kn) {
        const int s_base = kt * 32;

        // ---- S^T = K x Q^T (4 chained WMMAs)
        v8f st0 = vz, st1 = vz;
        st0 = wmma_bf16(kc[0], qb[0], st0);
        st0 = wmma_bf16(kc[1], qb[1], st0);
        st1 = wmma_bf16(kc[2], qb[0], st1);
        st1 = wmma_bf16(kc[3], qb[1], st1);

        // ---- issue ALL V loads for this tile now (consumed after softmax)
        V16 va[8];
#pragma unroll
        for (int cb = 0; cb < 8; ++cb) {
            const unsigned short* vrow =
                Vt + ((size_t)(b * VOUT + 16 * cb + ln)) * SEQ + s_base;
            va[cb].q[0] = *(const uint4*)(vrow + 16 * kh);
            va[cb].q[1] = *(const uint4*)(vrow + 16 * kh + 8);
        }
        // ---- next K tile in flight too
        if (kt + 1 < nkt) loadK(s_base + 32, kn);
        if (kt + 2 < nkt) {   // warm L2 two tiles ahead
            __builtin_prefetch(Km + (size_t)(b * SEQ + s_base + 64 + ln) * HEADD, 0, 1);
            __builtin_prefetch(Km + (size_t)(b * SEQ + s_base + 80 + ln) * HEADD, 0, 1);
        }

        // ---- scale + causal mask; D layout: row s, col t = ln
        float f0[8], f1[8];
#pragma unroll
        for (int r = 0; r < 8; ++r) {
            const int sl = s_base + r + 8 * kh;
            f0[r] = (sl      <= tg) ? st0[r] * SCALE : -1e30f;
            f1[r] = (sl + 16 <= tg) ? st1[r] * SCALE : -1e30f;
        }

        // ---- per-lane stats: in-register reduce over s, one SWAPX16
        float vmax = f0[0];
#pragma unroll
        for (int r = 1; r < 8; ++r) vmax = fmaxf(vmax, f0[r]);
#pragma unroll
        for (int r = 0; r < 8; ++r) vmax = fmaxf(vmax, f1[r]);
        vmax = fmaxf(vmax, xor16(vmax));

        const float mnew = fmaxf(mrun, vmax);
        const float al   = __expf(mrun - mnew);
        mrun = mnew;

        float p0[8], p1[8], rs = 0.f;
#pragma unroll
        for (int r = 0; r < 8; ++r) {
            p0[r] = __expf(f0[r] - mnew);
            p1[r] = __expf(f1[r] - mnew);
            rs += p0[r] + p1[r];
        }
        rs += xor16(rs);
        lrun = lrun * al + rs;

        // ---- build P^T as B operand (own half + partner half via SWAPX16)
        unsigned short lowh[8], highh[8];
#pragma unroll
        for (int r = 0; r < 8; ++r) {
            const float send = kh ? p0[r] : p1[r];
            const float recv = xor16(send);
            const float lo = kh ? recv : p0[r];
            const float hi = kh ? p1[r] : recv;
            lowh[r]  = f2bf_bits(lo);
            highh[r] = f2bf_bits(hi);
        }
        V16 pb;
        pb.q[0] = make_uint4(pk2(lowh[0], lowh[1]),   pk2(lowh[2], lowh[3]),
                             pk2(lowh[4], lowh[5]),   pk2(lowh[6], lowh[7]));
        pb.q[1] = make_uint4(pk2(highh[0], highh[1]), pk2(highh[2], highh[3]),
                             pk2(highh[4], highh[5]), pk2(highh[6], highh[7]));

        // ---- O^T += V^T x P^T
#pragma unroll
        for (int cb = 0; cb < 8; ++cb) {
            v8f c = accT[cb] * al;
            accT[cb] = wmma_bf16(va[cb], pb, c);
        }
    };

    // ---- main loop: K double-buffered with constant roles (no copies)
    V16 kA[4], kB[4];
    loadK(0, kA);
    int kt = 0;
    for (;;) {
        step(kt, kA, kB);
        if (++kt >= nkt) break;
        step(kt, kB, kA);
        if (++kt >= nkt) break;
    }

    // ---- epilogue: O^T tile: row col = 16cb + r + 8kh, col t = ln
    const float invl = 1.0f / lrun;
    float* orow = out + ((size_t)(b * SEQ + t_base + ln)) * VOUT;
#pragma unroll
    for (int cb = 0; cb < 8; ++cb)
#pragma unroll
        for (int r = 0; r < 8; ++r)
            orow[16 * cb + r + 8 * kh] = accT[cb][r] * invl;
}

// ---------------------------------------------------------------------------
// out = O1 - lam * O2  (float4 vectorized)
// ---------------------------------------------------------------------------
__global__ __launch_bounds__(256) void combine_kernel(const float* __restrict__ O,
                                                      const float* __restrict__ lamp,
                                                      float* __restrict__ out)
{
    const size_t i = ((size_t)blockIdx.x * 256 + threadIdx.x) * 4;
    const float lam = lamp[0];
    const float4 a = *(const float4*)(O + i);
    const float4 c = *(const float4*)(O + (size_t)NTOK * VOUT + i);
    float4 o;
    o.x = a.x - lam * c.x; o.y = a.y - lam * c.y;
    o.z = a.z - lam * c.z; o.w = a.w - lam * c.w;
    *(float4*)(out + i) = o;
}

// ---------------------------------------------------------------------------
extern "C" void kernel_launch(void* const* d_in, const int* in_sizes, int n_in,
                              void* d_out, int out_size, void* d_ws, size_t ws_size,
                              hipStream_t stream)
{
    (void)in_sizes; (void)n_in; (void)out_size; (void)ws_size;

    const float* x   = (const float*)d_in[0];
    const float* Wq1 = (const float*)d_in[1];
    const float* Wk1 = (const float*)d_in[2];
    const float* Wq2 = (const float*)d_in[3];
    const float* Wk2 = (const float*)d_in[4];
    const float* Wv  = (const float*)d_in[5];
    const float* lq1 = (const float*)d_in[6];
    const float* lk1 = (const float*)d_in[7];
    const float* lq2 = (const float*)d_in[8];
    const float* lk2 = (const float*)d_in[9];
    const int*  lidx = (const int*)d_in[10];

    unsigned short* q1 = (unsigned short*)d_ws;
    unsigned short* k1 = q1 + (size_t)NTOK * HEADD;
    unsigned short* q2 = k1 + (size_t)NTOK * HEADD;
    unsigned short* k2 = q2 + (size_t)NTOK * HEADD;
    unsigned short* Vt = k2 + (size_t)NTOK * HEADD;
    float* lam         = (float*)(Vt + (size_t)BATCH * VOUT * SEQ);
    unsigned short* xb = (unsigned short*)(lam + 4);           // 16B aligned
    unsigned short* Wt = xb + (size_t)NTOK * EMBD;
    float* Obuf        = (float*)(Wt + (size_t)NCOL * EMBD);   // 2 x [B*T][128]
    float* out = (float*)d_out;

    xcvt_kernel<<<(NTOK * EMBD / 4) / 256, 256, 0, stream>>>(x, xb);
    wcvt_kernel<<<dim3(NCOL, EMBD / 256), 256, 0, stream>>>(Wq1, Wk1, Wq2, Wk2, Wv, Wt);

    const int nwavesP = (NTOK / 32) * 6;      // 32-token x 64-col per wave
    proj_wmma_kernel<<<nwavesP / 4, 128, 0, stream>>>(xb, Wt, q1, k1, q2, k2, Vt);

    lam_kernel<<<1, 32, 0, stream>>>(lq1, lk1, lq2, lk2, lidx, lam);

    attn_kernel<<<dim3((NTOK / 16) / 4, 2), 128, 0, stream>>>(q1, k1, q2, k2, Vt, Obuf);
    combine_kernel<<<(NTOK * VOUT / 4) / 256, 256, 0, stream>>>(Obuf, lam, out);
}
